// TemporalGATGRU_62886911148786
// MI455X (gfx1250) — matrix operand order repaired and checked by reading, not compiled, not run on hardware
//
#include <hip/hip_runtime.h>
#include <hip/hip_bf16.h>

#define NEGS 0.2f
#define LNEPS 1e-5f

typedef __attribute__((ext_vector_type(16))) __bf16 v16bf;
typedef __attribute__((ext_vector_type(8)))  float  v8f;

__device__ __forceinline__ float lrelu(float f) { return f > 0.f ? f : NEGS * f; }

// order-preserving float<->uint encoding for atomicMax-based segment max
__device__ __forceinline__ unsigned fenc(float f) {
    unsigned u = __float_as_uint(f);
    return (u & 0x80000000u) ? ~u : (u | 0x80000000u);
}
__device__ __forceinline__ float fdec(unsigned u) {
    return (u & 0x80000000u) ? __uint_as_float(u ^ 0x80000000u) : __uint_as_float(~u);
}

__device__ __forceinline__ void edge_sd(const int* __restrict__ ei, int E, int e, int& s, int& d) {
    if (e < E) { s = ei[e]; d = ei[E + e]; }
    else       { s = e - E; d = e - E; }   // appended self loops
}

// ---------------------------------------------------------------------------
// WMMA bf16 GEMM:  C[M,Nout] = A[M,K] @ W[Nout,K]^T (+ bias[Nout])
// One wave computes a 16 x (16*NT) stripe: A fragment loaded once per K-step
// and reused across NT independent accumulators (independent back-to-back
// v_wmma issues, no RAW hazard stalls). 4 waves per block along M.
// A frag (16x32 bf16): lane hi=l>>4, m=l&15; element j -> K = j + 16*hi
// B frag (32x16 bf16): lane n=l&15, hi=l>>4; element j -> K = j + 16*hi
// D (16x16 f32): vgpr r -> m = r + 8*hi, n = l&15
// ---------------------------------------------------------------------------
template <int KDIM, int NT>
__global__ void gemm_wmma_bf16(const float* __restrict__ A, const float* __restrict__ W,
                               const float* __restrict__ bias, float* __restrict__ C,
                               int M, int Nout) {
    const int lane = threadIdx.x & 31;
    const int wave = threadIdx.x >> 5;
    const int tileM = (blockIdx.x * 4 + wave) * 16;
    const int tileN = blockIdx.y * (16 * NT);
    if (tileM >= M) return;
    const int hi = lane >> 4;
    const int lo = lane & 15;

    const float* Aptr = A + (size_t)(tileM + lo) * KDIM;
    const float* Wptr[NT];
#pragma unroll
    for (int t = 0; t < NT; ++t)
        Wptr[t] = W + (size_t)(tileN + 16 * t + lo) * KDIM;

    v8f zero = {};
    v8f acc[NT];
#pragma unroll
    for (int t = 0; t < NT; ++t) acc[t] = zero;

    for (int k0 = 0; k0 < KDIM; k0 += 32) {
        v16bf a;
#pragma unroll
        for (int p = 0; p < 8; ++p) {
            const int koff = k0 + 2 * p + 16 * hi;
            const float2 f2 = *(const float2*)(Aptr + koff);
            a[2 * p]     = (__bf16)f2.x;
            a[2 * p + 1] = (__bf16)f2.y;
        }
#pragma unroll
        for (int t = 0; t < NT; ++t) {
            v16bf b;
#pragma unroll
            for (int j = 0; j < 16; j += 4) {
                const float4 f4 = *(const float4*)(Wptr[t] + k0 + 16 * hi + j);
                b[j]     = (__bf16)f4.x;
                b[j + 1] = (__bf16)f4.y;
                b[j + 2] = (__bf16)f4.z;
                b[j + 3] = (__bf16)f4.w;
            }
            acc[t] = __builtin_amdgcn_wmma_f32_16x16x32_bf16(false, a, false, b,
                                                             (short)0, acc[t], false, false);
        }
    }

#pragma unroll
    for (int t = 0; t < NT; ++t) {
        const int n = tileN + 16 * t + lo;
        const float bv = bias ? bias[n] : 0.f;
#pragma unroll
        for (int r = 0; r < 8; ++r) {
            const int m = tileM + r + 8 * hi;
            C[(size_t)m * Nout + n] = acc[t][r] + bv;
        }
    }
}

// ---------------------------------------------------------------------------
__global__ void fill_f32(float* __restrict__ p, float v, long long n) {
    long long i = (long long)blockIdx.x * blockDim.x + threadIdx.x;
    if (i < n) p[i] = v;
}

// per (node, head) attention scores: a_src[n,h] = <H[n,h,:], att_src[h,:]>
template <int H>
__global__ void att_scores(const float* __restrict__ Hm, const float* __restrict__ as,
                           const float* __restrict__ ad, float* __restrict__ osrc,
                           float* __restrict__ odst, int N) {
    int tid = blockIdx.x * blockDim.x + threadIdx.x;
    if (tid >= N * H) return;
    int h = tid % H;
    const float* hp = Hm + (size_t)tid * 64;
    const float* ap = as + h * 64;
    const float* bp = ad + h * 64;
    float s0 = 0.f, s1 = 0.f;
#pragma unroll 8
    for (int c = 0; c < 64; ++c) { float v = hp[c]; s0 += v * ap[c]; s1 += v * bp[c]; }
    osrc[tid] = s0;
    odst[tid] = s1;
}

template <int H>
__global__ void edge_logit(const int* __restrict__ ei, int E, int Ep,
                           const float* __restrict__ asrc, const float* __restrict__ adst,
                           float* __restrict__ el, unsigned* __restrict__ mbuf) {
    int e = blockIdx.x * blockDim.x + threadIdx.x;
    if (e >= Ep) return;
    int s, d; edge_sd(ei, E, e, s, d);
#pragma unroll
    for (int h = 0; h < H; ++h) {
        float v = lrelu(asrc[s * H + h] + adst[d * H + h]);
        el[(size_t)e * H + h] = v;
        atomicMax(&mbuf[d * H + h], fenc(v));
    }
}

template <int H>
__global__ void edge_exp(const int* __restrict__ ei, int E, int Ep,
                         float* __restrict__ el, const unsigned* __restrict__ mbuf,
                         float* __restrict__ sbuf) {
    int e = blockIdx.x * blockDim.x + threadIdx.x;
    if (e >= Ep) return;
    int s, d; edge_sd(ei, E, e, s, d);
#pragma unroll
    for (int h = 0; h < H; ++h) {
        float w = expf(el[(size_t)e * H + h] - fdec(mbuf[d * H + h]));
        el[(size_t)e * H + h] = w;
        atomicAdd(&sbuf[d * H + h], w);
    }
}

// one thread per (edge, head, 4-feature chunk)
template <int H>
__global__ void edge_scatter(const int* __restrict__ ei, int E, int Ep,
                             const float* __restrict__ el, const float* __restrict__ sbuf,
                             const float* __restrict__ hsrc, float* __restrict__ outp) {
    long long tid = (long long)blockIdx.x * blockDim.x + threadIdx.x;
    long long total = (long long)Ep * H * 16;
    if (tid >= total) return;
    int c4 = (int)(tid & 15);
    long long t = tid >> 4;
    int h = (int)(t % H);
    int e = (int)(t / H);
    int s, d; edge_sd(ei, E, e, s, d);
    float alpha = el[(size_t)e * H + h] / sbuf[d * H + h];
    const float4 v = *(const float4*)(hsrc + ((size_t)s * H + h) * 64 + c4 * 4);
    float* op = outp + ((size_t)d * H + h) * 64 + c4 * 4;
    atomicAdd(op + 0, alpha * v.x);
    atomicAdd(op + 1, alpha * v.y);
    atomicAdd(op + 2, alpha * v.z);
    atomicAdd(op + 3, alpha * v.w);
}

// in-place: p[i] = lrelu(p[i] + b[i % F])
__global__ void bias_lrelu(float* __restrict__ p, const float* __restrict__ b,
                           int F, long long n) {
    long long i = (long long)blockIdx.x * blockDim.x + threadIdx.x;
    if (i >= n) return;
    p[i] = lrelu(p[i] + b[(int)(i % F)]);
}

// h = lrelu(o2 + b2); LN(h)*gamma+beta + R  -> hm   (biased variance, torch-style)
__global__ void ln_residual(const float* __restrict__ o2, const float* __restrict__ b2,
                            const float* __restrict__ gamma, const float* __restrict__ beta,
                            const float* __restrict__ R, float* __restrict__ hm, int N) {
    int n = blockIdx.x * blockDim.x + threadIdx.x;
    if (n >= N) return;
    const float* op = o2 + (size_t)n * 64;
    float sum = 0.f, ssq = 0.f;
#pragma unroll 8
    for (int c = 0; c < 64; ++c) { float t = lrelu(op[c] + b2[c]); sum += t; ssq += t * t; }
    const float mu = sum * (1.f / 64.f);
    const float var = ssq * (1.f / 64.f) - mu * mu;
    const float rinv = rsqrtf(var + LNEPS);
#pragma unroll 8
    for (int c = 0; c < 64; ++c) {
        float t = lrelu(op[c] + b2[c]);
        hm[(size_t)n * 64 + c] = (t - mu) * rinv * gamma[c] + beta[c] + R[(size_t)n * 64 + c];
    }
}

// single-step GRU (h0 = 0) fused with fc head (3 classes)
__global__ void gru_fc(const float* __restrict__ gi, const float* __restrict__ bhh,
                       const float* __restrict__ fcw, const float* __restrict__ fcb,
                       float* __restrict__ out, int N) {
    int n = blockIdx.x * blockDim.x + threadIdx.x;
    if (n >= N) return;
    const float* g = gi + (size_t)n * 192;
    float a0 = 0.f, a1 = 0.f, a2 = 0.f;
#pragma unroll 4
    for (int c = 0; c < 64; ++c) {
        float r  = 1.f / (1.f + expf(-(g[c] + bhh[c])));
        float z  = 1.f / (1.f + expf(-(g[64 + c] + bhh[64 + c])));
        float ng = tanhf(g[128 + c] + r * bhh[128 + c]);
        float hy = (1.f - z) * ng;
        a0 += hy * fcw[c];
        a1 += hy * fcw[64 + c];
        a2 += hy * fcw[128 + c];
    }
    out[(size_t)n * 3 + 0] = a0 + fcb[0];
    out[(size_t)n * 3 + 1] = a1 + fcb[1];
    out[(size_t)n * 3 + 2] = a2 + fcb[2];
}

// ---------------------------------------------------------------------------
extern "C" void kernel_launch(void* const* d_in, const int* in_sizes, int n_in,
                              void* d_out, int out_size, void* d_ws, size_t ws_size,
                              hipStream_t stream) {
    const float* x        = (const float*)d_in[0];
    const int*   ei       = (const int*)d_in[1];
    const float* W1       = (const float*)d_in[2];
    const float* att_src1 = (const float*)d_in[3];
    const float* att_dst1 = (const float*)d_in[4];
    const float* b1       = (const float*)d_in[5];
    const float* W2       = (const float*)d_in[6];
    const float* att_src2 = (const float*)d_in[7];
    const float* att_dst2 = (const float*)d_in[8];
    const float* b2       = (const float*)d_in[9];
    const float* lin_w    = (const float*)d_in[10];
    const float* lin_b    = (const float*)d_in[11];
    const float* gamma    = (const float*)d_in[12];
    const float* beta     = (const float*)d_in[13];
    const float* w_ih     = (const float*)d_in[14];
    // d_in[15] = w_hh: unused, h0 == 0
    const float* b_ih     = (const float*)d_in[16];
    const float* b_hh     = (const float*)d_in[17];
    const float* fc_w     = (const float*)d_in[18];
    const float* fc_b     = (const float*)d_in[19];
    float* out = (float*)d_out;
    (void)n_in; (void)out_size; (void)ws_size;

    const int N  = in_sizes[0] / 256;
    const int E  = in_sizes[1] / 2;
    const int Ep = E + N;

    // workspace carving (floats)
    float* ws = (float*)d_ws;
    size_t off = 0;
    float* H1   = ws + off; off += (size_t)N * 256;  // gat1 features; reused for GRU gi
    float* X2   = ws + off; off += (size_t)N * 256;  // gat1 aggregation -> lrelu'd input to gat2
    float* EL   = ws + off; off += (size_t)Ep * 4;   // per-edge logits / weights
    float* ASRC = ws + off; off += (size_t)N * 4;
    float* ADST = ws + off; off += (size_t)N * 4;
    float* MBUF = ws + off; off += (size_t)N * 4;    // encoded max (uint)
    float* SBUF = ws + off; off += (size_t)N * 4;    // softmax denom
    float* H2   = ws + off; off += (size_t)N * 64;   // gat2 features; reused for hmid
    float* O2   = ws + off; off += (size_t)N * 64;   // gat2 aggregation
    float* R    = ws + off; off += (size_t)N * 64;   // residual x@lin_w^T + lin_b
    float* GI   = H1;                                // N*192 <= N*256

    const int TB = 256;
    auto blocks = [](long long n, int tb) { return (unsigned)((n + tb - 1) / tb); };

    // ---------------- GAT layer 1 (H=4, C=64) ----------------
    fill_f32<<<blocks((long long)N * 256, TB), TB, 0, stream>>>(X2, 0.f, (long long)N * 256);
    fill_f32<<<blocks((long long)N * 8, TB), TB, 0, stream>>>(MBUF, 0.f, (long long)N * 8); // MBUF+SBUF

    gemm_wmma_bf16<256, 4><<<dim3((N + 63) / 64, 4), 128, 0, stream>>>(x, W1, nullptr, H1, N, 256);
    att_scores<4><<<blocks((long long)N * 4, TB), TB, 0, stream>>>(H1, att_src1, att_dst1, ASRC, ADST, N);
    edge_logit<4><<<blocks(Ep, TB), TB, 0, stream>>>(ei, E, Ep, ASRC, ADST, EL, (unsigned*)MBUF);
    edge_exp<4><<<blocks(Ep, TB), TB, 0, stream>>>(ei, E, Ep, EL, (const unsigned*)MBUF, SBUF);
    edge_scatter<4><<<blocks((long long)Ep * 64, TB), TB, 0, stream>>>(ei, E, Ep, EL, SBUF, H1, X2);
    bias_lrelu<<<blocks((long long)N * 256, TB), TB, 0, stream>>>(X2, b1, 256, (long long)N * 256);

    // ---------------- GAT layer 2 (H=1, C=64) ----------------
    fill_f32<<<blocks((long long)N * 64, TB), TB, 0, stream>>>(O2, 0.f, (long long)N * 64);
    fill_f32<<<blocks((long long)N * 8, TB), TB, 0, stream>>>(MBUF, 0.f, (long long)N * 8);

    gemm_wmma_bf16<256, 4><<<dim3((N + 63) / 64, 1), 128, 0, stream>>>(X2, W2, nullptr, H2, N, 64);
    att_scores<1><<<blocks(N, TB), TB, 0, stream>>>(H2, att_src2, att_dst2, ASRC, ADST, N);
    edge_logit<1><<<blocks(Ep, TB), TB, 0, stream>>>(ei, E, Ep, ASRC, ADST, EL, (unsigned*)MBUF);
    edge_exp<1><<<blocks(Ep, TB), TB, 0, stream>>>(ei, E, Ep, EL, (const unsigned*)MBUF, SBUF);
    edge_scatter<1><<<blocks((long long)Ep * 16, TB), TB, 0, stream>>>(ei, E, Ep, EL, SBUF, H2, O2);

    // ---------------- LN + residual + GRU + FC ----------------
    gemm_wmma_bf16<256, 4><<<dim3((N + 63) / 64, 1), 128, 0, stream>>>(x, lin_w, lin_b, R, N, 64);
    ln_residual<<<blocks(N, TB), TB, 0, stream>>>(O2, b2, gamma, beta, R, H2 /*hmid*/, N);
    gemm_wmma_bf16<64, 4><<<dim3((N + 63) / 64, 3), 128, 0, stream>>>(H2, w_ih, b_ih, GI, N, 192);
    gru_fc<<<blocks(N, TB), TB, 0, stream>>>(GI, b_hh, fc_w, fc_b, out, N);
}
